// FlashShellAttention_58145267253385
// MI455X (gfx1250) — matrix operand order repaired and checked by reading, not compiled, not run on hardware
//
#include <hip/hip_runtime.h>
#include <hip/hip_bf16.h>
#include <cstdint>
#include <cstddef>

#define NH     32
#define NKVH   8
#define DHEAD  128
#define HIDSZ  4096
#define BATCH  4
#define SEQ    1024
#define TOK    4096
#define NCACHE 16384
#define QKVN   6144          // NH*DHEAD + 2*NKVH*DHEAD
#define KVW    1024          // NKVH*DHEAD

typedef __attribute__((ext_vector_type(16))) __bf16 bf16x16;
typedef __attribute__((ext_vector_type(8)))  __bf16 bf16x8;
typedef __attribute__((ext_vector_type(4)))  __bf16 bf16x4;
typedef __attribute__((ext_vector_type(8)))  float  f32x8;

__device__ __forceinline__ bf16x16 join16(bf16x8 lo, bf16x8 hi) {
  return __builtin_shufflevector(lo, hi, 0,1,2,3,4,5,6,7,8,9,10,11,12,13,14,15);
}

// CDNA5 async copy: global memory -> LDS, 16B per lane, tracked by ASYNCcnt.
// LDS byte offset = low 32 bits of the generic pointer (aperture rule).
__device__ __forceinline__ void async_ld_b128(void* lds, const void* gptr) {
  asm volatile("global_load_async_to_lds_b128 %0, %1, off"
               :: "v"((uint32_t)(uintptr_t)lds), "v"(gptr)
               : "memory");
}
__device__ __forceinline__ void wait_async0() {
  asm volatile("s_wait_asynccnt 0x0" ::: "memory");
}

// ---------------------------------------------------------------------------
// fp32 -> bf16 bulk convert (vectorized, grid-stride). n4 = elements/4.
// ---------------------------------------------------------------------------
__global__ __launch_bounds__(256)
void cvt_f32_bf16(const float* __restrict__ s, __bf16* __restrict__ d, size_t n4) {
  for (size_t i = blockIdx.x * (size_t)blockDim.x + threadIdx.x; i < n4;
       i += (size_t)gridDim.x * blockDim.x) {
    const float4 v = reinterpret_cast<const float4*>(s)[i];
    bf16x4 o;
    o[0] = (__bf16)v.x; o[1] = (__bf16)v.y; o[2] = (__bf16)v.z; o[3] = (__bf16)v.w;
    reinterpret_cast<bf16x4*>(d)[i] = o;
  }
}

// ---------------------------------------------------------------------------
// C[M,N] = A[M,K](bf16) * B[N,K]^T(bf16) + bias[N], fp32 out.
// Block: 128x128 tile, 256 threads = 8 waves (2x4). Wave: 64x32 via 4x2 WMMA.
// ---------------------------------------------------------------------------
__global__ __launch_bounds__(256)
void gemm_bf16_nt(const __bf16* __restrict__ A, const __bf16* __restrict__ B,
                  const float* __restrict__ bias, float* __restrict__ C,
                  int M, int N, int K) {
  const int wave = threadIdx.x >> 5;
  const int lane = threadIdx.x & 31;
  const int half = lane >> 4;
  const int l16  = lane & 15;
  const int wm = wave >> 2, wn = wave & 3;
  const int row0 = blockIdx.x * 128 + wm * 64;
  const int col0 = blockIdx.y * 128 + wn * 32;

  f32x8 acc[4][2] = {};

  const __bf16* aptr[4];
  const __bf16* bptr[2];
#pragma unroll
  for (int i = 0; i < 4; i++)
    aptr[i] = A + (size_t)(row0 + i * 16 + l16) * K + half * 8;
#pragma unroll
  for (int j = 0; j < 2; j++)
    bptr[j] = B + (size_t)(col0 + j * 16 + l16) * K + half * 8;

  for (int k0 = 0; k0 < K; k0 += 32) {
    if (k0 + 256 < K) {
      __builtin_prefetch(aptr[0] + k0 + 256, 0, 3);   // WGP-scope prefetch
      __builtin_prefetch(bptr[0] + k0 + 256, 0, 3);
    }
    bf16x16 af[4], bfr[2];
#pragma unroll
    for (int i = 0; i < 4; i++) {
      bf16x8 lo = *reinterpret_cast<const bf16x8*>(aptr[i] + k0);
      bf16x8 hi = *reinterpret_cast<const bf16x8*>(aptr[i] + k0 + 16);
      af[i] = join16(lo, hi);
    }
#pragma unroll
    for (int j = 0; j < 2; j++) {
      bf16x8 lo = *reinterpret_cast<const bf16x8*>(bptr[j] + k0);
      bf16x8 hi = *reinterpret_cast<const bf16x8*>(bptr[j] + k0 + 16);
      bfr[j] = join16(lo, hi);
    }
#pragma unroll
    for (int i = 0; i < 4; i++)
#pragma unroll
      for (int j = 0; j < 2; j++)
        acc[i][j] = __builtin_amdgcn_wmma_f32_16x16x32_bf16(
            false, af[i], false, bfr[j], (short)0, acc[i][j], false, false);
  }

#pragma unroll
  for (int i = 0; i < 4; i++)
#pragma unroll
    for (int j = 0; j < 2; j++) {
      const int col = col0 + j * 16 + l16;
      const float bv = bias[col];
#pragma unroll
      for (int r = 0; r < 8; r++) {
        const int row = row0 + i * 16 + r + half * 8;
        C[(size_t)row * N + col] = acc[i][j][r] + bv;
      }
    }
}

// ---------------------------------------------------------------------------
// RoPE on q/k, emit bf16 q/k/v for attention, scatter fp32 k/v into caches.
// ---------------------------------------------------------------------------
__global__ __launch_bounds__(256)
void rope_scatter(const float* __restrict__ qkv, const float* __restrict__ cs,
                  const float* __restrict__ sn, const int* __restrict__ slots,
                  __bf16* __restrict__ qb, __bf16* __restrict__ kb,
                  __bf16* __restrict__ vb,
                  float* __restrict__ kc, float* __restrict__ vc) {
  const size_t NQ = (size_t)TOK * NH * 64;
  const size_t NK = (size_t)TOK * NKVH * 64;
  const size_t NV = (size_t)TOK * NKVH * DHEAD;
  const size_t total = NQ + NK + NV;
  for (size_t idx = blockIdx.x * (size_t)blockDim.x + threadIdx.x; idx < total;
       idx += (size_t)gridDim.x * blockDim.x) {
    if (idx < NQ) {
      const int d = (int)(idx & 63);
      const size_t th = idx >> 6;
      const int h = (int)(th % NH);
      const int t = (int)(th / NH);
      const float c = cs[(size_t)t * 64 + d], s = sn[(size_t)t * 64 + d];
      const float* r = qkv + (size_t)t * QKVN + h * DHEAD;
      const float x1 = r[d], x2 = r[d + 64];
      const size_t o = (size_t)t * HIDSZ + h * DHEAD + d;
      qb[o]      = (__bf16)(x1 * c - x2 * s);
      qb[o + 64] = (__bf16)(x2 * c + x1 * s);
    } else if (idx < NQ + NK) {
      const size_t rel = idx - NQ;
      const int d = (int)(rel & 63);
      const int g = (int)((rel >> 6) % NKVH);
      const int t = (int)(rel / (64 * NKVH));
      const float c = cs[(size_t)t * 64 + d], s = sn[(size_t)t * 64 + d];
      const float* r = qkv + (size_t)t * QKVN + HIDSZ + g * DHEAD;
      const float x1 = r[d], x2 = r[d + 64];
      const float y1 = x1 * c - x2 * s, y2 = x2 * c + x1 * s;
      const size_t o = (size_t)t * KVW + g * DHEAD + d;
      kb[o] = (__bf16)y1; kb[o + 64] = (__bf16)y2;
      const size_t co = (size_t)slots[t] * KVW + g * DHEAD + d;
      kc[co] = y1; kc[co + 64] = y2;
    } else {
      const size_t rel = idx - NQ - NK;
      const int d = (int)(rel & 127);
      const int g = (int)((rel >> 7) % NKVH);
      const int t = (int)(rel / (128 * NKVH));
      const float x = qkv[(size_t)t * QKVN + HIDSZ + KVW + g * DHEAD + d];
      vb[(size_t)t * KVW + g * DHEAD + d] = (__bf16)x;
      vc[(size_t)slots[t] * KVW + g * DHEAD + d] = x;
    }
  }
}

// ---------------------------------------------------------------------------
// Causal GQA flash attention, bf16 WMMA, fp32 accumulate / online softmax.
// K tile staged via GLOBAL_LOAD_ASYNC_TO_LDS_B128 (ASYNCcnt); V staged with an
// explicit transpose store (async copy cannot transpose).
// Grid: (S/128, NH, BATCH). 256 threads = 8 waves; wave owns 16 q rows.
// ---------------------------------------------------------------------------
__global__ __launch_bounds__(256)
void attn_fwd(const __bf16* __restrict__ Q, const __bf16* __restrict__ Kq,
              const __bf16* __restrict__ Vq, __bf16* __restrict__ O) {
  __shared__ __align__(32) __bf16 Ks[32][DHEAD + 8];   // pad: conflict-free rows
  __shared__ __align__(32) __bf16 Vs[DHEAD][40];       // transposed V, padded
  __shared__ __align__(32) __bf16 Ps[8][16][40];       // per-wave P staging

  const int qt = blockIdx.x, h = blockIdx.y, b = blockIdx.z;
  const int g = h >> 2;                                // GQA: 4 q heads / kv head
  const int wave = threadIdx.x >> 5, lane = threadIdx.x & 31;
  const int half = lane >> 4, l16 = lane & 15;
  const int q0 = qt * 128 + wave * 16;                 // wave's first q row in seq

  // Preload Q fragments (A operand) for all 4 K-steps over D=128.
  bf16x16 qf[4];
  {
    const __bf16* qp = Q + (size_t)(b * SEQ + q0 + l16) * HIDSZ + h * DHEAD;
#pragma unroll
    for (int kk = 0; kk < 4; kk++) {
      bf16x8 lo = *reinterpret_cast<const bf16x8*>(qp + kk * 32 + half * 8);
      bf16x8 hi = *reinterpret_cast<const bf16x8*>(qp + kk * 32 + half * 8 + 16);
      qf[kk] = join16(lo, hi);
    }
  }

  f32x8 oacc[8] = {};
  float mrow[8], lrow[8];
#pragma unroll
  for (int r = 0; r < 8; r++) { mrow[r] = -3.0e38f; lrow[r] = 0.f; }

  const float scale = 0.08838834764831845f;            // 1/sqrt(128)
  const int njt = qt * 4 + 4;                          // causal k-tile count

  for (int j = 0; j < njt; j++) {
    // Stage K tile (row-major) via async DMA to LDS; V via transpose store.
    {
      const int key = threadIdx.x >> 3;
      const int d0  = (threadIdx.x & 7) * 16;
      const size_t t = (size_t)(b * SEQ + j * 32 + key);
      const __bf16* ksrc = Kq + t * KVW + g * DHEAD + d0;
      async_ld_b128(&Ks[key][d0],     ksrc);           // ASYNCcnt-tracked
      async_ld_b128(&Ks[key][d0 + 8], ksrc + 8);
      const __bf16* vsrc = Vq + t * KVW + g * DHEAD + d0;
      bf16x8 v0 = *reinterpret_cast<const bf16x8*>(vsrc);
      bf16x8 v1 = *reinterpret_cast<const bf16x8*>(vsrc + 8);
#pragma unroll
      for (int e = 0; e < 8; e++) { Vs[d0 + e][key] = v0[e]; Vs[d0 + 8 + e][key] = v1[e]; }
      wait_async0();                                   // s_wait_asynccnt 0
    }
    __syncthreads();

    if (j * 32 <= q0 + 15) {   // wave-uniform causal skip (barriers still hit)
      // ---- S = Q K^T : 8 WMMAs over D=128 contraction, 2 key subtiles ----
      f32x8 sc[2] = {};
#pragma unroll
      for (int kk = 0; kk < 4; kk++)
#pragma unroll
        for (int sn2 = 0; sn2 < 2; sn2++) {
          const __bf16* p = &Ks[sn2 * 16 + l16][kk * 32 + half * 8];
          bf16x16 bfv = join16(*reinterpret_cast<const bf16x8*>(p),
                               *reinterpret_cast<const bf16x8*>(p + 16));
          sc[sn2] = __builtin_amdgcn_wmma_f32_16x16x32_bf16(
              false, qf[kk], false, bfv, (short)0, sc[sn2], false, false);
        }

      // ---- online softmax in WMMA C layout (VGPR r = row r+8*half) ----
#pragma unroll
      for (int r = 0; r < 8; r++) {
        const int qrow = q0 + r + half * 8;
        float s0 = sc[0][r] * scale; if (j * 32 + l16      > qrow) s0 = -1e30f;
        float s1 = sc[1][r] * scale; if (j * 32 + 16 + l16 > qrow) s1 = -1e30f;
        float mloc = fmaxf(s0, s1);
#pragma unroll
        for (int off = 8; off > 0; off >>= 1)
          mloc = fmaxf(mloc, __shfl_xor(mloc, off, 32));
        const float mnew = fmaxf(mrow[r], mloc);
        const float corr = __expf(mrow[r] - mnew);
        const float p0 = __expf(s0 - mnew);
        const float p1 = __expf(s1 - mnew);
        float ps = p0 + p1;
#pragma unroll
        for (int off = 8; off > 0; off >>= 1) ps += __shfl_xor(ps, off, 32);
        lrow[r] = lrow[r] * corr + ps;
        mrow[r] = mnew;
#pragma unroll
        for (int n = 0; n < 8; n++) oacc[n][r] *= corr;
        Ps[wave][r + half * 8][l16]      = (__bf16)p0;    // re-layout via LDS
        Ps[wave][r + half * 8][l16 + 16] = (__bf16)p1;    // (same-wave DS in order)
      }

      // ---- O += P V : read P back as A operand, V^T rows as B operand ----
      bf16x16 pf;
      {
        const __bf16* p = &Ps[wave][l16][half * 8];
        pf = join16(*reinterpret_cast<const bf16x8*>(p),
                    *reinterpret_cast<const bf16x8*>(p + 16));
      }
#pragma unroll
      for (int n = 0; n < 8; n++) {
        const __bf16* vp = &Vs[n * 16 + l16][half * 8];
        bf16x16 bv = join16(*reinterpret_cast<const bf16x8*>(vp),
                            *reinterpret_cast<const bf16x8*>(vp + 16));
        oacc[n] = __builtin_amdgcn_wmma_f32_16x16x32_bf16(
            false, pf, false, bv, (short)0, oacc[n], false, false);
      }
    }
    __syncthreads();
  }

  float inv[8];
#pragma unroll
  for (int r = 0; r < 8; r++) inv[r] = 1.0f / lrow[r];
#pragma unroll
  for (int n = 0; n < 8; n++)
#pragma unroll
    for (int r = 0; r < 8; r++) {
      const size_t t = (size_t)(b * SEQ + q0 + r + half * 8);
      O[t * HIDSZ + h * DHEAD + n * 16 + l16] = (__bf16)(oacc[n][r] * inv[r]);
    }
}

// ---------------------------------------------------------------------------
extern "C" void kernel_launch(void* const* d_in, const int* in_sizes, int n_in,
                              void* d_out, int out_size, void* d_ws, size_t ws_size,
                              hipStream_t stream) {
  (void)in_sizes; (void)n_in; (void)out_size; (void)ws_size;
  const float* hs    = (const float*)d_in[0];
  const float* cosp  = (const float*)d_in[1];
  const float* sinp  = (const float*)d_in[2];
  const float* w_qkv = (const float*)d_in[3];
  const float* b_qkv = (const float*)d_in[4];
  const float* w_o   = (const float*)d_in[5];
  const float* b_o   = (const float*)d_in[6];
  const float* kc_in = (const float*)d_in[7];
  const float* vc_in = (const float*)d_in[8];
  const int*   slots = (const int*)d_in[9];

  float* out    = (float*)d_out;                       // [TOK, HIDSZ]
  float* kc_out = out + (size_t)TOK * HIDSZ;           // [NCACHE, KVW]
  float* vc_out = kc_out + (size_t)NCACHE * KVW;       // [NCACHE, KVW]

  // Workspace layout (bytes): ~302 MB total.
  char* ws = (char*)d_ws;
  __bf16* hs_b   = (__bf16*)ws;  ws += (size_t)TOK  * HIDSZ * 2;
  __bf16* wqkv_b = (__bf16*)ws;  ws += (size_t)QKVN * HIDSZ * 2;
  __bf16* wo_b   = (__bf16*)ws;  ws += (size_t)HIDSZ * HIDSZ * 2;
  float*  qkv_f  = (float*)ws;   ws += (size_t)TOK * QKVN * 4;
  __bf16* q_b    = (__bf16*)ws;  ws += (size_t)TOK * HIDSZ * 2;
  __bf16* k_b    = (__bf16*)ws;  ws += (size_t)TOK * KVW * 2;
  __bf16* v_b    = (__bf16*)ws;  ws += (size_t)TOK * KVW * 2;
  __bf16* attn_b = (__bf16*)ws;  ws += (size_t)TOK * HIDSZ * 2;

  // 1) seed cache outputs with the input caches (scatter overwrites slots).
  hipMemcpyAsync(kc_out, kc_in, (size_t)NCACHE * KVW * 4,
                 hipMemcpyDeviceToDevice, stream);
  hipMemcpyAsync(vc_out, vc_in, (size_t)NCACHE * KVW * 4,
                 hipMemcpyDeviceToDevice, stream);

  // 2) fp32 -> bf16 operand conversion (bandwidth-bound, one pass each).
  cvt_f32_bf16<<<4096, 256, 0, stream>>>(hs,    hs_b,   (size_t)TOK * HIDSZ / 4);
  cvt_f32_bf16<<<4096, 256, 0, stream>>>(w_qkv, wqkv_b, (size_t)QKVN * HIDSZ / 4);
  cvt_f32_bf16<<<4096, 256, 0, stream>>>(w_o,   wo_b,   (size_t)HIDSZ * HIDSZ / 4);

  // 3) QKV projection: [4096x4096] x [6144x4096]^T + bias -> fp32.
  gemm_bf16_nt<<<dim3(TOK / 128, QKVN / 128), 256, 0, stream>>>(
      hs_b, wqkv_b, b_qkv, qkv_f, TOK, QKVN, HIDSZ);

  // 4) RoPE + bf16 repack + fp32 cache scatter.
  rope_scatter<<<4096, 256, 0, stream>>>(qkv_f, cosp, sinp, slots,
                                         q_b, k_b, v_b, kc_out, vc_out);

  // 5) causal GQA flash attention.
  attn_fwd<<<dim3(SEQ / 128, NH, BATCH), 256, 0, stream>>>(q_b, k_b, v_b, attn_b);

  // 6) output projection: [4096x4096] x [4096x4096]^T + bias -> d_out.
  gemm_bf16_nt<<<dim3(TOK / 128, HIDSZ / 128), 256, 0, stream>>>(
      attn_b, wo_b, b_o, out, TOK, HIDSZ, HIDSZ);
}
